// GNNModel_1898375545371
// MI455X (gfx1250) — matrix operand order repaired and checked by reading, not compiled, not run on hardware
//
#include <hip/hip_runtime.h>
#include <math.h>

typedef __attribute__((ext_vector_type(16))) _Float16 v16h;
typedef __attribute__((ext_vector_type(8)))  _Float16 v8h;
typedef __attribute__((ext_vector_type(4)))  _Float16 v4h;
typedef __attribute__((ext_vector_type(8)))  float    v8f;

#define NNODES 100000
#define NEDGES 400000
#define NGRAPH 1024

// ---------------------------------------------------------------- utilities

__global__ void zero_f32(float* __restrict__ p, int n) {
  int stride = gridDim.x * blockDim.x;
  for (int i = blockIdx.x * blockDim.x + threadIdx.x; i < n; i += stride) p[i] = 0.0f;
}

__global__ void deg_count(const int* __restrict__ dst, float* __restrict__ deg, int E) {
  int e = blockIdx.x * blockDim.x + threadIdx.x;
  if (e < E) atomicAdd(&deg[dst[e]], 1.0f);
}

__global__ void dinv_k(const float* __restrict__ deg, float* __restrict__ dinv, int n) {
  int i = blockIdx.x * blockDim.x + threadIdx.x;
  if (i < n) dinv[i] = rsqrtf(deg[i] + 1.0f);
}

__global__ void enorm_k(const int* __restrict__ src, const int* __restrict__ dst,
                        const float* __restrict__ dinv, float* __restrict__ en, int E) {
  int e = blockIdx.x * blockDim.x + threadIdx.x;
  if (e < E) en[e] = dinv[src[e]] * dinv[dst[e]];
}

// x [N,90] f32 -> xh [N,96] f16 (zero pad K so WMMA K-steps of 32 divide evenly)
__global__ void pad_x_h(const float* __restrict__ x, _Float16* __restrict__ xh, int total) {
  int idx = blockIdx.x * blockDim.x + threadIdx.x;
  if (idx >= total) return;
  int i = idx / 96, f = idx - i * 96;
  xh[idx] = (f < 90) ? (_Float16)x[i * 90 + f] : (_Float16)0.0f;
}

// W [K,Nout] f32 row-major -> Wt [Nout,Kp] f16 (col-major B, K zero-padded to Kp)
__global__ void cvt_w(const float* __restrict__ W, _Float16* __restrict__ Wt,
                      int K, int Kp, int Nout, int total) {
  int idx = blockIdx.x * blockDim.x + threadIdx.x;
  if (idx >= total) return;
  int n = idx / Kp, k = idx - n * Kp;
  Wt[idx] = (k < K) ? (_Float16)W[(size_t)k * Nout + n] : (_Float16)0.0f;
}

__global__ void f32_to_f16(const float* __restrict__ in, _Float16* __restrict__ out, int n) {
  int i = blockIdx.x * blockDim.x + threadIdx.x;
  if (i < n) out[i] = (_Float16)in[i];
}

// ------------------------------------------------ WMMA GEMM (f16 in, f32 acc)
// C[M,Nout] = A[M,K] * B[K,Nout] (+bias,+relu). A: row-major f16 [M,lda].
// Bt: col-major f16 [Nout,K] (row n = column n of B) -> contiguous K per lane.
// One wave computes a 16x32 output tile (two 16x16 WMMA sharing the A fragment).
// Fragment layouts used here (A and B share contiguous K order):
//   A 16x32: lane<16 M=lane elems 0-15=K0-15; lane>=16 M=lane-16,
//            elems 0-15=K16-31.
//   B 32x16: lane<16 N=lane, elem j = K j; lane>=16 N=lane-16, elem j = K16+j.
//   C/D 16x16 f32: lane<16 N=lane, VGPR r -> M=r; lane>=16 -> M=8+r.
__global__ __launch_bounds__(256)
void wmma_gemm_h(const _Float16* __restrict__ A, int lda,
                 const _Float16* __restrict__ Bt,
                 float* __restrict__ C, int ldc,
                 int M, int K, int Nout,
                 const float* __restrict__ bias, int do_relu) {
  const int wave = threadIdx.x >> 5;
  const int lane = threadIdx.x & 31;
  const int nT = Nout >> 5;                 // 32-wide N tiles
  const int tot = (M >> 4) * nT;
  const int tile = blockIdx.x * 8 + wave;
  if (tile >= tot) return;                  // wave-uniform: EXEC all-1s at WMMA
  const int tm = tile / nT;
  const int tn = tile - tm * nT;
  const int l16 = lane & 15;
  const bool hi = lane >= 16;
  const int row  = tm * 16 + l16;
  const int col0 = tn * 32 + l16;
  const int col1 = col0 + 16;
  const int abase = hi ? 16 : 0;
  const int bbase = hi ? 16 : 0;
  const _Float16* pA  = A  + (size_t)row  * lda + abase;
  const _Float16* pB0 = Bt + (size_t)col0 * K  + bbase;
  const _Float16* pB1 = Bt + (size_t)col1 * K  + bbase;

  v8f acc0 = {}, acc1 = {};
  for (int k0 = 0; k0 < K; k0 += 32) {
    v8h alo = *reinterpret_cast<const v8h*>(pA + k0);        // K = k0+abase .. +7
    v8h ahi = *reinterpret_cast<const v8h*>(pA + k0 + 8);    // K = k0+abase+8..+15
    v16h av = __builtin_shufflevector(alo, ahi,
        0, 1, 2, 3, 4, 5, 6, 7, 8, 9, 10, 11, 12, 13, 14, 15);
    v16h bv0 = *reinterpret_cast<const v16h*>(pB0 + k0);     // 16 contiguous K
    v16h bv1 = *reinterpret_cast<const v16h*>(pB1 + k0);
    if (k0 + 32 < K) __builtin_prefetch(pA + k0 + 32, 0, 1); // global_prefetch_b8
    acc0 = __builtin_amdgcn_wmma_f32_16x16x32_f16(
        false, av, false, bv0, (short)0, acc0, false, false);
    acc1 = __builtin_amdgcn_wmma_f32_16x16x32_f16(
        false, av, false, bv1, (short)0, acc1, false, false);
  }

  const float b0 = bias ? bias[col0] : 0.0f;
  const float b1 = bias ? bias[col1] : 0.0f;
  const int r0 = tm * 16 + (hi ? 8 : 0);
#pragma unroll
  for (int r = 0; r < 8; ++r) {
    float v0 = acc0[r] + b0;
    float v1 = acc1[r] + b1;
    if (do_relu) { v0 = fmaxf(v0, 0.0f); v1 = fmaxf(v1, 0.0f); }
    C[(size_t)(r0 + r) * ldc + col0] = v0;
    C[(size_t)(r0 + r) * ldc + col1] = v1;
  }
}

// ------------------------------------------------------------- GCN scatter

__global__ void edge_scatter(const int* __restrict__ src, const int* __restrict__ dst,
                             const float* __restrict__ en, const float* __restrict__ h,
                             float* __restrict__ agg, int F, int total) {
  int idx = blockIdx.x * blockDim.x + threadIdx.x;
  if (idx >= total) return;
  int fq = F >> 2;
  int e = idx / fq, fc = (idx - e * fq) << 2;
  int s = src[e], d = dst[e];
  float nrm = en[e];
  float4 hv = *reinterpret_cast<const float4*>(h + (size_t)s * F + fc);
  float* p = agg + (size_t)d * F + fc;
  atomicAdd(p + 0, hv.x * nrm);
  atomicAdd(p + 1, hv.y * nrm);
  atomicAdd(p + 2, hv.z * nrm);
  atomicAdd(p + 3, hv.w * nrm);
}

// x_out = relu(agg + h*dinv^2 + b); writes f32 (scatter/pool path) + f16 (next GEMM)
__global__ void finalize_k(const float* __restrict__ agg, const float* __restrict__ h,
                           const float* __restrict__ dinv, const float* __restrict__ b,
                           float* __restrict__ xf, _Float16* __restrict__ xh,
                           int F, int total) {
  int idx = blockIdx.x * blockDim.x + threadIdx.x;
  if (idx >= total) return;
  int fq = F >> 2;
  int i = idx / fq, fc = (idx - i * fq) << 2;
  float di = dinv[i];
  float sl = di * di;
  float4 a  = *reinterpret_cast<const float4*>(agg + (size_t)i * F + fc);
  float4 hv = *reinterpret_cast<const float4*>(h   + (size_t)i * F + fc);
  float4 bb = *reinterpret_cast<const float4*>(b + fc);
  float4 v;
  v.x = fmaxf(a.x + hv.x * sl + bb.x, 0.0f);
  v.y = fmaxf(a.y + hv.y * sl + bb.y, 0.0f);
  v.z = fmaxf(a.z + hv.z * sl + bb.z, 0.0f);
  v.w = fmaxf(a.w + hv.w * sl + bb.w, 0.0f);
  *reinterpret_cast<float4*>(xf + (size_t)i * F + fc) = v;
  v4h hh;
  hh[0] = (_Float16)v.x; hh[1] = (_Float16)v.y;
  hh[2] = (_Float16)v.z; hh[3] = (_Float16)v.w;
  *reinterpret_cast<v4h*>(xh + (size_t)i * F + fc) = hh;
}

// ---------------------------------------------------------------- pooling

__device__ __forceinline__ void atomicMaxPos(float* a, float v) {
  // int-bitcast ordering valid for non-negative floats (post-ReLU values)
  atomicMax(reinterpret_cast<int*>(a), __float_as_int(v));
}

__global__ void cnt_k(const int* __restrict__ batch, float* __restrict__ cnt, int n) {
  int i = blockIdx.x * blockDim.x + threadIdx.x;
  if (i < n) atomicAdd(&cnt[batch[i]], 1.0f);
}

__global__ void pool_k(const float* __restrict__ x, const int* __restrict__ batch,
                       float* __restrict__ gmax, float* __restrict__ gsum, int total) {
  int idx = blockIdx.x * blockDim.x + threadIdx.x;
  if (idx >= total) return;
  int i = idx >> 6, fc = (idx & 63) << 2;
  int g = batch[i];
  float4 v = *reinterpret_cast<const float4*>(x + (size_t)i * 256 + fc);
  float* pm = gmax + (size_t)g * 256 + fc;
  float* ps = gsum + (size_t)g * 256 + fc;
  atomicMaxPos(pm + 0, v.x); atomicAdd(ps + 0, v.x);
  atomicMaxPos(pm + 1, v.y); atomicAdd(ps + 1, v.y);
  atomicMaxPos(pm + 2, v.z); atomicAdd(ps + 2, v.z);
  atomicMaxPos(pm + 3, v.w); atomicAdd(ps + 3, v.w);
}

__global__ void pool_final(const float* __restrict__ gmax, const float* __restrict__ gsum,
                           const float* __restrict__ cnt, float* __restrict__ pooled, int total) {
  int idx = blockIdx.x * blockDim.x + threadIdx.x;
  if (idx >= total) return;
  int g = idx >> 8, f = idx & 255;
  pooled[(size_t)g * 512 + f] = gmax[idx];
  pooled[(size_t)g * 512 + 256 + f] = gsum[idx] / cnt[g];
}

// ------------------------------------------------------------------- MLP

// d = concat(h1-h2, h2-h1) written directly as f16 GEMM input
__global__ void diff_k(const float* __restrict__ p1, const float* __restrict__ p2,
                       _Float16* __restrict__ d, int total) {
  int idx = blockIdx.x * blockDim.x + threadIdx.x;
  if (idx >= total) return;
  int g = idx >> 9, f = idx & 511;
  float a = p1[idx], b = p2[idx];
  d[(size_t)g * 1024 + f]       = (_Float16)(a - b);
  d[(size_t)g * 1024 + 512 + f] = (_Float16)(b - a);
}

__global__ void fc3_tanh(const float* __restrict__ t2, const float* __restrict__ W,
                         const float* __restrict__ b, float* __restrict__ out, int G) {
  int g = blockIdx.x * blockDim.x + threadIdx.x;
  if (g >= G) return;
  float acc = b[0];
#pragma unroll 8
  for (int j = 0; j < 256; ++j) acc += t2[(size_t)g * 256 + j] * W[j];
  out[g] = tanhf(acc);
}

// ------------------------------------------------------------------- host

static inline void launch_gemm(const _Float16* A, int lda, const _Float16* Bt,
                               float* C, int ldc, int M, int K, int Nout,
                               const float* bias, int relu, hipStream_t s) {
  int tiles = (M / 16) * (Nout / 32);
  int blocks = (tiles + 7) / 8;  // 8 waves / block
  wmma_gemm_h<<<blocks, 256, 0, s>>>(A, lda, Bt, C, ldc, M, K, Nout, bias, relu);
}

static void encoder(const float* x, const int* ei, const int* batch,
                    const _Float16* W1t, const float* b1, const _Float16* W2t, const float* b2,
                    const _Float16* W3t, const float* b3, const _Float16* W4t, const float* b4,
                    float* xA, _Float16* xh, float* hbuf, float* agg,
                    float* deg, float* dinv, float* enorm,
                    float* cnt, float* gsum, float* gmax, float* pooled, hipStream_t stream) {
  const int* src = ei;
  const int* dst = ei + NEDGES;
  const int T = 256;

  zero_f32<<<1024, T, 0, stream>>>(deg, NNODES);
  deg_count<<<(NEDGES + T - 1) / T, T, 0, stream>>>(dst, deg, NEDGES);
  dinv_k<<<(NNODES + T - 1) / T, T, 0, stream>>>(deg, dinv, NNODES);
  enorm_k<<<(NEDGES + T - 1) / T, T, 0, stream>>>(src, dst, dinv, enorm, NEDGES);
  pad_x_h<<<(NNODES * 96 + T - 1) / T, T, 0, stream>>>(x, xh, NNODES * 96);

  // layer 1: 96 -> 128
  launch_gemm(xh, 96, W1t, hbuf, 128, NNODES, 96, 128, nullptr, 0, stream);
  zero_f32<<<4096, T, 0, stream>>>(agg, NNODES * 128);
  edge_scatter<<<(NEDGES * 32 + T - 1) / T, T, 0, stream>>>(src, dst, enorm, hbuf, agg, 128, NEDGES * 32);
  finalize_k<<<(NNODES * 32 + T - 1) / T, T, 0, stream>>>(agg, hbuf, dinv, b1, xA, xh, 128, NNODES * 32);

  // layers 2-4: ->256 ->256 ->256
  const _Float16* Ws[3] = {W2t, W3t, W4t};
  const float*    bs[3] = {b2, b3, b4};
  int Kin = 128;
  for (int l = 0; l < 3; ++l) {
    launch_gemm(xh, Kin, Ws[l], hbuf, 256, NNODES, Kin, 256, nullptr, 0, stream);
    zero_f32<<<8192, T, 0, stream>>>(agg, NNODES * 256);
    edge_scatter<<<(NEDGES * 64 + T - 1) / T, T, 0, stream>>>(src, dst, enorm, hbuf, agg, 256, NEDGES * 64);
    finalize_k<<<(NNODES * 64 + T - 1) / T, T, 0, stream>>>(agg, hbuf, dinv, bs[l], xA, xh, 256, NNODES * 64);
    Kin = 256;
  }

  // pooling (f32 path)
  zero_f32<<<1024, T, 0, stream>>>(gmax, NGRAPH * 256);
  zero_f32<<<1024, T, 0, stream>>>(gsum, NGRAPH * 256);
  zero_f32<<<16, T, 0, stream>>>(cnt, NGRAPH);
  cnt_k<<<(NNODES + T - 1) / T, T, 0, stream>>>(batch, cnt, NNODES);
  pool_k<<<(NNODES * 64 + T - 1) / T, T, 0, stream>>>(xA, batch, gmax, gsum, NNODES * 64);
  pool_final<<<(NGRAPH * 256 + T - 1) / T, T, 0, stream>>>(gmax, gsum, cnt, pooled, NGRAPH * 256);
}

extern "C" void kernel_launch(void* const* d_in, const int* in_sizes, int n_in,
                              void* d_out, int out_size, void* d_ws, size_t ws_size,
                              hipStream_t stream) {
  const float* x1     = (const float*)d_in[0];
  const int*   ei1    = (const int*)d_in[1];
  const int*   batch1 = (const int*)d_in[2];
  const float* x2     = (const float*)d_in[3];
  const int*   ei2    = (const int*)d_in[4];
  const int*   batch2 = (const int*)d_in[5];
  const float* W1 = (const float*)d_in[6];   const float* b1 = (const float*)d_in[7];
  const float* W2 = (const float*)d_in[8];   const float* b2 = (const float*)d_in[9];
  const float* W3 = (const float*)d_in[10];  const float* b3 = (const float*)d_in[11];
  const float* W4 = (const float*)d_in[12];  const float* b4 = (const float*)d_in[13];
  const float* fc1W = (const float*)d_in[14]; const float* fc1b = (const float*)d_in[15];
  const float* fc2W = (const float*)d_in[16]; const float* fc2b = (const float*)d_in[17];
  const float* fc3W = (const float*)d_in[18]; const float* fc3b = (const float*)d_in[19];

  char* ws = (char*)d_ws;
  size_t off = 0;
  auto alloc = [&](size_t bytes) -> char* {
    char* p = ws + off;
    off = (off + bytes + 255) & ~(size_t)255;
    return p;
  };
  float*    xA    = (float*)   alloc((size_t)NNODES * 256 * 4);
  float*    hbuf  = (float*)   alloc((size_t)NNODES * 256 * 4);
  float*    agg   = (float*)   alloc((size_t)NNODES * 256 * 4);
  _Float16* xh    = (_Float16*)alloc((size_t)NNODES * 256 * 2);
  float*    deg   = (float*)   alloc((size_t)NNODES * 4);
  float*    dinv  = (float*)   alloc((size_t)NNODES * 4);
  float*    enorm = (float*)   alloc((size_t)NEDGES * 4);
  _Float16* W1t   = (_Float16*)alloc(128 * 96 * 2);
  _Float16* W2t   = (_Float16*)alloc(256 * 128 * 2);
  _Float16* W3t   = (_Float16*)alloc(256 * 256 * 2);
  _Float16* W4t   = (_Float16*)alloc(256 * 256 * 2);
  _Float16* fc1Wt = (_Float16*)alloc(512 * 1024 * 2);
  _Float16* fc2Wt = (_Float16*)alloc(256 * 512 * 2);
  float*    pool1 = (float*)   alloc((size_t)NGRAPH * 512 * 4);
  float*    pool2 = (float*)   alloc((size_t)NGRAPH * 512 * 4);
  float*    cnt   = (float*)   alloc((size_t)NGRAPH * 4);
  float*    gsum  = (float*)   alloc((size_t)NGRAPH * 256 * 4);
  float*    gmax  = (float*)   alloc((size_t)NGRAPH * 256 * 4);
  _Float16* dbufh = (_Float16*)alloc((size_t)NGRAPH * 1024 * 2);
  float*    t1    = (float*)   alloc((size_t)NGRAPH * 512 * 4);
  _Float16* t1h   = (_Float16*)alloc((size_t)NGRAPH * 512 * 2);
  float*    t2    = (float*)   alloc((size_t)NGRAPH * 256 * 4);
  (void)in_sizes; (void)n_in; (void)out_size; (void)ws_size;

  const int T = 256;
  // one-time weight transpose/convert to col-major f16
  cvt_w<<<(128 * 96 + T - 1) / T, T, 0, stream>>>(W1, W1t, 90, 96, 128, 128 * 96);
  cvt_w<<<(256 * 128 + T - 1) / T, T, 0, stream>>>(W2, W2t, 128, 128, 256, 256 * 128);
  cvt_w<<<(256 * 256 + T - 1) / T, T, 0, stream>>>(W3, W3t, 256, 256, 256, 256 * 256);
  cvt_w<<<(256 * 256 + T - 1) / T, T, 0, stream>>>(W4, W4t, 256, 256, 256, 256 * 256);
  cvt_w<<<(512 * 1024 + T - 1) / T, T, 0, stream>>>(fc1W, fc1Wt, 1024, 1024, 512, 512 * 1024);
  cvt_w<<<(256 * 512 + T - 1) / T, T, 0, stream>>>(fc2W, fc2Wt, 512, 512, 256, 256 * 512);

  encoder(x1, ei1, batch1, W1t, b1, W2t, b2, W3t, b3, W4t, b4,
          xA, xh, hbuf, agg, deg, dinv, enorm, cnt, gsum, gmax, pool1, stream);
  encoder(x2, ei2, batch2, W1t, b1, W2t, b2, W3t, b3, W4t, b4,
          xA, xh, hbuf, agg, deg, dinv, enorm, cnt, gsum, gmax, pool2, stream);

  // MLP head (f16 WMMA path)
  diff_k<<<(NGRAPH * 512 + T - 1) / T, T, 0, stream>>>(pool1, pool2, dbufh, NGRAPH * 512);
  launch_gemm(dbufh, 1024, fc1Wt, t1, 512, NGRAPH, 1024, 512, fc1b, 1, stream);
  f32_to_f16<<<(NGRAPH * 512 + T - 1) / T, T, 0, stream>>>(t1, t1h, NGRAPH * 512);
  launch_gemm(t1h, 512, fc2Wt, t2, 256, NGRAPH, 512, 256, fc2b, 1, stream);
  fc3_tanh<<<(NGRAPH + T - 1) / T, T, 0, stream>>>(t2, fc3W, fc3b, (float*)d_out, NGRAPH);
}
